// nonConsecutiveConvLayer5_56796647522678
// MI455X (gfx1250) — matrix operand order repaired and checked by reading, not compiled, not run on hardware
//
#include <hip/hip_runtime.h>
#include <hip/hip_bf16.h>

// CDNA5 (gfx1250) wave32 WMMA types
typedef __attribute__((ext_vector_type(16))) __bf16 v16bf;
typedef __attribute__((ext_vector_type(8)))  float  v8f;
// Matches the async-to-LDS builtin's parameter type exactly:
typedef int async_v4i __attribute__((vector_size(16)));
typedef __attribute__((address_space(3))) async_v4i* lds_v4i_ptr;

#define B_   128
#define L_   128
#define D_   1024
#define F_   1024
#define W_   5
#define KB   64             // K block staged in LDS per iteration
#define NKB  (D_ / KB)      // 16
#define FT   16             // F-columns per workgroup
#define NPW  (W_ * FT)      // 80 proj columns per WG
#define APAD (KB + 8)       // 72 elems -> 144B row stride, conflict-free, 16B aligned

#define A_ELEMS (L_ * APAD)          // 9216  (18432 B)
#define B_ELEMS (W_ * FT * APAD)     // 5760  (11520 B)
#define BUF_ELEMS (A_ELEMS + B_ELEMS)        // 14976 elems = 29952 B
// total static LDS = 59904 B (< 64KB); proj dump (40960 B) aliases it

#define HAS_ASYNC_LDS (__has_builtin(__builtin_amdgcn_global_load_async_to_lds_b128) && \
                       __has_builtin(__builtin_amdgcn_s_wait_asynccnt))

union FragBF16 {
    uint4 u[2];
    v16bf v;
};

// ---------------------------------------------------------------------------
// Prep kernel 1: x fp32 -> bf16 (same [B,L,D] layout), 4 elems / thread-iter
// ---------------------------------------------------------------------------
__global__ void cdna5_convert_x_kernel(const float4* __restrict__ in,
                                       uint2* __restrict__ outv, int n4)
{
    for (int i = blockIdx.x * blockDim.x + threadIdx.x; i < n4;
         i += gridDim.x * blockDim.x) {
        const float4 v = in[i];
        union { __bf16 h[4]; uint2 u; } p;
        p.h[0] = (__bf16)v.x; p.h[1] = (__bf16)v.y;
        p.h[2] = (__bf16)v.z; p.h[3] = (__bf16)v.w;
        outv[i] = p.u;
    }
}

// ---------------------------------------------------------------------------
// Prep kernel 2: Ws [w][d][f] fp32 -> wt [w][f][d] bf16 (64x64 LDS transpose)
// ---------------------------------------------------------------------------
__global__ void cdna5_transpose_ws_kernel(const float* __restrict__ Ws,
                                          __bf16* __restrict__ wt)
{
    __shared__ __bf16 tile[64 * 65];
    const int w  = blockIdx.z;
    const int d0 = blockIdx.x * 64;
    const int f0 = blockIdx.y * 64;
    for (int idx = threadIdx.x; idx < 64 * 64; idx += 256) {
        const int r = idx >> 6, c = idx & 63;    // r: d, c: f (coalesced read)
        tile[r * 65 + c] = (__bf16)Ws[((size_t)w * D_ + d0 + r) * F_ + f0 + c];
    }
    __syncthreads();
    for (int idx = threadIdx.x; idx < 64 * 64; idx += 256) {
        const int c = idx >> 6, r = idx & 63;    // output row f0+c, col d0+r (r fast)
        wt[((size_t)w * F_ + f0 + c) * D_ + d0 + r] = tile[r * 65 + c];
    }
}

// ---------------------------------------------------------------------------
// Main fused kernel: bf16 sources, async-to-LDS double-buffered staging
// ---------------------------------------------------------------------------
__global__ __launch_bounds__(256)
void nonConsecutiveConvLayer5_56796647522678_kernel(
        const __bf16* __restrict__ xq,   // [B, L, D] bf16
        const __bf16* __restrict__ wq,   // [W, F, D] bf16 (transposed)
        const float*  __restrict__ bias, // [F]
        float* __restrict__ out)         // [B, F]
{
    const int tid  = threadIdx.x;
    const int b    = blockIdx.y;
    const int f0   = blockIdx.x * FT;
    const int wv   = tid >> 5;        // wave id == M-tile (16 timesteps)
    const int lane = tid & 31;
    const int n    = lane & 15;
    const int half = lane >> 4;

    __shared__ __align__(16) __bf16 smem[2 * BUF_ELEMS];
    float* sP = (float*)smem;         // aliased proj [L_][NPW], used after GEMM

    v8f acc[W_];
#pragma unroll
    for (int w = 0; w < W_; ++w) {
        v8f z = {0.f, 0.f, 0.f, 0.f, 0.f, 0.f, 0.f, 0.f};
        acc[w] = z;
    }

    const __bf16* xb = xq + (size_t)b * L_ * D_;

    // Stage one K-block: pure 16B copies global->LDS (async DMA when available)
    auto stage = [&](int kb, int buf) {
        __bf16* sA = smem + buf * BUF_ELEMS;
        __bf16* sB = sA + A_ELEMS;
        const int kbase = kb * KB;
        // A tile: 128 rows x 64 k = 1024 16B-chunks (4 per thread)
#pragma unroll
        for (int i = 0; i < 4; ++i) {
            const int c  = tid + i * 256;
            const int l  = c >> 3, cc = c & 7;
            const __bf16* g = xb + (size_t)l * D_ + kbase + cc * 8;
            __bf16*       s = sA + l * APAD + cc * 8;
#if HAS_ASYNC_LDS
            __builtin_amdgcn_global_load_async_to_lds_b128(
                (async_v4i*)g, (lds_v4i_ptr)s, 0, 0);
#else
            *(uint4*)s = *(const uint4*)g;
#endif
        }
        // B tile: 80 rows x 64 k = 640 16B-chunks
#pragma unroll
        for (int i = 0; i < 3; ++i) {
            const int c = tid + i * 256;
            if (c < (W_ * FT * KB) / 8) {
                const int row = c >> 3, cc = c & 7;       // row = w*16 + fl
                const int w = row >> 4, fl = row & 15;
                const __bf16* g = wq + ((size_t)w * F_ + f0 + fl) * D_ + kbase + cc * 8;
                __bf16*       s = sB + row * APAD + cc * 8;
#if HAS_ASYNC_LDS
                __builtin_amdgcn_global_load_async_to_lds_b128(
                    (async_v4i*)g, (lds_v4i_ptr)s, 0, 0);
#else
                *(uint4*)s = *(const uint4*)g;
#endif
            }
        }
    };

    stage(0, 0);
#if HAS_ASYNC_LDS
    __builtin_amdgcn_s_wait_asynccnt(0);
#endif
    __syncthreads();

    for (int kb = 0; kb < NKB; ++kb) {
        // async-fill the other buffer while this one feeds the WMMA pipe
        if (kb + 1 < NKB) stage(kb + 1, (kb + 1) & 1);

        const __bf16* sA = smem + (kb & 1) * BUF_ELEMS;
        const __bf16* sB = sA + A_ELEMS;

#pragma unroll
        for (int ks = 0; ks < KB / 32; ++ks) {
            const int k0 = ks * 32;
            FragBF16 fa;
            {
                // A 16x32 bf16: lane half 0 -> K {0..7,16..23}, half 1 -> {8..15,24..31}
                const __bf16* pa = sA + (wv * 16 + n) * APAD + k0 + half * 8;
                fa.u[0] = *(const uint4*)(pa);
                fa.u[1] = *(const uint4*)(pa + 16);
            }
            FragBF16 fb[W_];
#pragma unroll
            for (int w = 0; w < W_; ++w) {
                // B 32x16 bf16: lane half 0 -> K {0..7,16..23}, half 1 -> {8..15,24..31}
                const __bf16* pb = sB + (w * FT + n) * APAD + k0 + half * 8;
                fb[w].u[0] = *(const uint4*)(pb);
                fb[w].u[1] = *(const uint4*)(pb + 16);
            }
#pragma unroll
            for (int w = 0; w < W_; ++w) {
                acc[w] = __builtin_amdgcn_wmma_f32_16x16x32_bf16(
                    false, fa.v, false, fb[w].v, (short)0, acc[w], false, false);
            }
        }
#if HAS_ASYNC_LDS
        __builtin_amdgcn_s_wait_asynccnt(0);
#endif
        __syncthreads();
    }

    // ---- dump proj tiles to LDS (aliases staging buffers; all reads done) ----
#pragma unroll
    for (int w = 0; w < W_; ++w) {
#pragma unroll
        for (int r = 0; r < 8; ++r) {
            const int m = r + half * 8;           // C layout: VGPR r -> row m / m+8
            const int l = wv * 16 + m;
            sP[l * NPW + w * FT + n] = acc[w][r];
        }
    }
    __syncthreads();

    // ---- fused recurrence over timesteps (16 lanes, one per f column) ----
    if (tid < FT) {
        float h0 = 0.f, h1 = 0.f, h2 = 0.f, h3 = 0.f, h4 = 0.f;
        for (int l = 0; l < L_; ++l) {
            const float* p = sP + l * NPW + tid;
            const float p0 = p[0 * FT], p1 = p[1 * FT], p2 = p[2 * FT],
                        p3 = p[3 * FT], p4 = p[4 * FT];
            h4 = fmaxf(h3 + p4, h4);
            h3 = fmaxf(h2 + p3, h3);
            h2 = fmaxf(h1 + p2, h2);
            h1 = fmaxf(h0 + p1, h1);
            h0 = fmaxf(p0,      h0);
        }
        out[(size_t)b * F_ + f0 + tid] = tanhf(h4 + bias[f0 + tid]);
    }
}

// ---------------------------------------------------------------------------
// Fallback fused kernel (fp32 sources, converts during staging) — used only
// when the workspace is too small for the bf16 copies.
// ---------------------------------------------------------------------------
__global__ __launch_bounds__(256)
void nonConsecutiveConvLayer5_fallback_kernel(
        const float* __restrict__ x, const float* __restrict__ Ws,
        const float* __restrict__ bias, float* __restrict__ out)
{
    const int tid  = threadIdx.x;
    const int b    = blockIdx.y;
    const int f0   = blockIdx.x * FT;
    const int wv   = tid >> 5;
    const int lane = tid & 31;
    const int n    = lane & 15;
    const int half = lane >> 4;

    __shared__ __align__(16) __bf16 smem[2 * BUF_ELEMS];
    float* sP = (float*)smem;

    v8f acc[W_];
#pragma unroll
    for (int w = 0; w < W_; ++w) {
        v8f z = {0.f, 0.f, 0.f, 0.f, 0.f, 0.f, 0.f, 0.f};
        acc[w] = z;
    }

    const float* xb = x + (size_t)b * L_ * D_;

    auto stage = [&](int kb, int buf) {
        __bf16* sA = smem + buf * BUF_ELEMS;
        __bf16* sB = sA + A_ELEMS;
        const int kbase = kb * KB;
        for (int idx = tid; idx < L_ * KB; idx += 256) {
            const int l = idx >> 6, k = idx & (KB - 1);
            sA[l * APAD + k] = (__bf16)xb[(size_t)l * D_ + kbase + k];
        }
        for (int idx = tid; idx < W_ * FT * KB; idx += 256) {
            const int fl = idx & (FT - 1);
            const int k  = (idx >> 4) & (KB - 1);
            const int w  = idx >> 10;
            sB[(w * FT + fl) * APAD + k] =
                (__bf16)Ws[((size_t)w * D_ + kbase + k) * F_ + f0 + fl];
        }
    };

    stage(0, 0);
    __syncthreads();

    for (int kb = 0; kb < NKB; ++kb) {
        if (kb + 1 < NKB) stage(kb + 1, (kb + 1) & 1);
        const __bf16* sA = smem + (kb & 1) * BUF_ELEMS;
        const __bf16* sB = sA + A_ELEMS;
#pragma unroll
        for (int ks = 0; ks < KB / 32; ++ks) {
            const int k0 = ks * 32;
            FragBF16 fa;
            const __bf16* pa = sA + (wv * 16 + n) * APAD + k0 + half * 8;
            fa.u[0] = *(const uint4*)(pa);
            fa.u[1] = *(const uint4*)(pa + 16);
            FragBF16 fb[W_];
#pragma unroll
            for (int w = 0; w < W_; ++w) {
                const __bf16* pb = sB + (w * FT + n) * APAD + k0 + half * 8;
                fb[w].u[0] = *(const uint4*)(pb);
                fb[w].u[1] = *(const uint4*)(pb + 16);
            }
#pragma unroll
            for (int w = 0; w < W_; ++w) {
                acc[w] = __builtin_amdgcn_wmma_f32_16x16x32_bf16(
                    false, fa.v, false, fb[w].v, (short)0, acc[w], false, false);
            }
        }
        __syncthreads();
    }

#pragma unroll
    for (int w = 0; w < W_; ++w) {
#pragma unroll
        for (int r = 0; r < 8; ++r) {
            const int m = r + half * 8;
            sP[(wv * 16 + m) * NPW + w * FT + n] = acc[w][r];
        }
    }
    __syncthreads();

    if (tid < FT) {
        float h0 = 0.f, h1 = 0.f, h2 = 0.f, h3 = 0.f, h4 = 0.f;
        for (int l = 0; l < L_; ++l) {
            const float* p = sP + l * NPW + tid;
            const float p0 = p[0 * FT], p1 = p[1 * FT], p2 = p[2 * FT],
                        p3 = p[3 * FT], p4 = p[4 * FT];
            h4 = fmaxf(h3 + p4, h4);
            h3 = fmaxf(h2 + p3, h3);
            h2 = fmaxf(h1 + p2, h2);
            h1 = fmaxf(h0 + p1, h1);
            h0 = fmaxf(p0,      h0);
        }
        out[(size_t)b * F_ + f0 + tid] = tanhf(h4 + bias[f0 + tid]);
    }
}

extern "C" void kernel_launch(void* const* d_in, const int* in_sizes, int n_in,
                              void* d_out, int out_size, void* d_ws, size_t ws_size,
                              hipStream_t stream) {
    const float* x    = (const float*)d_in[0]; // [128,128,1024]
    const float* Ws   = (const float*)d_in[1]; // [5,1024,1024]
    const float* bias = (const float*)d_in[2]; // [1024]
    float* out        = (float*)d_out;         // [128,1024]

    const size_t xq_elems = (size_t)B_ * L_ * D_;   // 16,777,216
    const size_t wq_elems = (size_t)W_ * D_ * F_;   //  5,242,880
    const size_t need     = (xq_elems + wq_elems) * sizeof(__bf16); // ~42 MB

    dim3 grid(F_ / FT, B_);   // 64 x 128 workgroups
    dim3 block(256);          // 8 wave32

    if (ws_size >= need) {
        __bf16* xq = (__bf16*)d_ws;
        __bf16* wq = xq + xq_elems;
        cdna5_convert_x_kernel<<<4096, 256, 0, stream>>>(
            (const float4*)x, (uint2*)xq, (int)(xq_elems / 4));
        cdna5_transpose_ws_kernel<<<dim3(D_ / 64, F_ / 64, W_), 256, 0, stream>>>(Ws, wq);
        nonConsecutiveConvLayer5_56796647522678_kernel<<<grid, block, 0, stream>>>(
            xq, wq, bias, out);
    } else {
        nonConsecutiveConvLayer5_fallback_kernel<<<grid, block, 0, stream>>>(
            x, Ws, bias, out);
    }
}
